// Fact_tt_Sam_21191368638972
// MI455X (gfx1250) — matrix operand order repaired and checked by reading, not compiled, not run on hardware
//
#include <hip/hip_runtime.h>
#include <hip/hip_bf16.h>
#include <math.h>

// ---------------------------------------------------------------------------
// MI455X (gfx1250) implementation: all dense contractions via
// v_wmma_f32_16x16x32_f16 (wave32); A-tiles staged with CDNA5 async
// global->LDS copies (ASYNCcnt) and double-buffered LDS.
// Conv3d lowered to im2col + GEMM.
// ---------------------------------------------------------------------------

typedef _Float16 half_t;
typedef __attribute__((ext_vector_type(16))) _Float16 v16h;
typedef __attribute__((ext_vector_type(8)))  _Float16 h8;
typedef __attribute__((ext_vector_type(8)))  float    v8f;
typedef __attribute__((ext_vector_type(4)))  unsigned int u32x4;
typedef __attribute__((ext_vector_type(2)))  unsigned int u32x2;

#define DIMC   768
#define HEADS  12
#define HDIM   64
#define WIN    14
#define ACH    192
#define RLO    32
#define DDEP   5
#define HWDIM  64
#define T_TOK  (DDEP * HWDIM * HWDIM)      /* 20480 image tokens            */
#define NWIN   125                         /* 5 imgs * 5*5 windows          */
#define NP     196                         /* tokens per window             */
#define NPAD   224                         /* NP padded to K%32==0          */
#define TW_TOK (NWIN * NP)                 /* 24500 windowed tokens         */
#define NBH    (NWIN * HEADS)              /* 1500 (window,head) batches    */
#define KCONV  (ACH * 27)                  /* 5184 im2col K                 */

__device__ __forceinline__ float gelu_f(float v) {
  return 0.5f * v * (1.0f + erff(v * 0.70710678118654752f));
}

// CDNA5 async global->LDS 128-bit copy (per-lane), tracked by ASYNCcnt.
__device__ __forceinline__ void async_b128(void* lds_dst, const void* gsrc) {
  // generic LDS pointer truncated to 32 bits == wave-relative LDS offset
  unsigned lds = (unsigned)(unsigned long long)lds_dst;
  asm volatile("global_load_async_to_lds_b128 %0, %1, off"
               :: "v"(lds), "v"(gsrc) : "memory");
}

// ---------------------------------------------------------------------------
// Generic batched WMMA GEMM:  C[M,N] = alpha * A[M,K] @ B[K,N] (+bias)(gelu)
//                                      (+resid)   -> f32 and/or f16 outputs
// A: f16 row-major, lda == K.  B: f16 row-major [K,N], ldb == N.
// Block: 256 threads = 8 wave32 waves.  Tile BM x BN, K step 32, K%32==0.
// A tile: async global->LDS, double buffered. B tile: transposed ds stores.
// ---------------------------------------------------------------------------
template <int BM, int BN, int WR, int WC, bool GELU>
__global__ __launch_bounds__(256) void wmma_gemm(
    const half_t* __restrict__ A, const half_t* __restrict__ B,
    const float* __restrict__ bias, const float* resid, int ldr,
    float alpha, float* Cf, half_t* Ch, int ldc,
    int M, int N, int K,
    long long strideA, long long strideB, long long strideC) {
  constexpr int FM = BM / (16 * WR);
  constexpr int FN = BN / (16 * WC);
  constexpr int ACHUNKS = BM * 4 / 256;     // per-thread b128 copies per tile
  alignas(16) __shared__ half_t sA[2][BM][40];   // [m][k], +8 pad halves
  alignas(16) __shared__ half_t sB[2][BN][40];   // transposed: [n][k], +8 pad

  const int tid  = threadIdx.x;
  const int lane = tid & 31;
  const int wave = tid >> 5;
  const int wr   = wave % WR;
  const int wc   = wave / WR;
  const int l16  = lane & 15;
  const int g    = lane >> 4;
  const int bm0  = blockIdx.y * BM;
  const int bn0  = blockIdx.x * BN;

  const half_t* Ab = A + (size_t)((long long)blockIdx.z * strideA);
  const half_t* Bb = B + (size_t)((long long)blockIdx.z * strideB);

  const bool fullM = (bm0 + BM <= M);
  const bool fullN = (bn0 + BN <= N);
  const int  nt    = K >> 5;

  v8f acc[FM][FN];
#pragma unroll
  for (int i = 0; i < FM; ++i)
#pragma unroll
    for (int j = 0; j < FN; ++j)
#pragma unroll
      for (int r = 0; r < 8; ++r) acc[i][j][r] = 0.0f;

  union U16 { v16h v; h8 h[2]; };

  auto stageA = [&](int t, int buf) {
    const int k0 = t << 5;
    if (fullM) {
#pragma unroll
      for (int cid = tid; cid < BM * 4; cid += 256) {
        const int row = cid >> 2;
        const int c8  = (cid & 3) * 8;
        async_b128(&sA[buf][row][c8], Ab + (size_t)(bm0 + row) * K + k0 + c8);
      }
    } else {
#pragma unroll
      for (int cid = tid; cid < BM * 4; cid += 256) {
        const int row = cid >> 2;
        const int c8  = (cid & 3) * 8;
        u32x4 val = {0u, 0u, 0u, 0u};
        if (bm0 + row < M)
          val = *(const u32x4*)(Ab + (size_t)(bm0 + row) * K + k0 + c8);
        *(u32x4*)&sA[buf][row][c8] = val;
      }
    }
  };

  auto stageB = [&](int t, int buf) {
    const int k0 = t << 5;
    if (fullN) {
#pragma unroll
      for (int cid = tid; cid < (BN / 4) * 32; cid += 256) {
        const int kr = cid / (BN / 4);
        const int n0 = (cid % (BN / 4)) * 4;
        union { u32x2 u; half_t h[4]; } cv;
        cv.u = *(const u32x2*)(Bb + (size_t)(k0 + kr) * N + bn0 + n0);
        sB[buf][n0 + 0][kr] = cv.h[0];
        sB[buf][n0 + 1][kr] = cv.h[1];
        sB[buf][n0 + 2][kr] = cv.h[2];
        sB[buf][n0 + 3][kr] = cv.h[3];
      }
    } else {
#pragma unroll
      for (int cid = tid; cid < (BN / 4) * 32; cid += 256) {
        const int kr = cid / (BN / 4);
        const int n0 = (cid % (BN / 4)) * 4;
        const int gn = bn0 + n0;
        const half_t* src = Bb + (size_t)(k0 + kr) * N + gn;
        half_t v0 = 0.0f, v1 = 0.0f, v2 = 0.0f, v3 = 0.0f;
        if (gn     < N) v0 = src[0];
        if (gn + 1 < N) v1 = src[1];
        if (gn + 2 < N) v2 = src[2];
        if (gn + 3 < N) v3 = src[3];
        sB[buf][n0 + 0][kr] = v0;
        sB[buf][n0 + 1][kr] = v1;
        sB[buf][n0 + 2][kr] = v2;
        sB[buf][n0 + 3][kr] = v3;
      }
    }
  };

  stageA(0, 0);                 // prologue: async copies for tile 0
  for (int t = 0; t < nt; ++t) {
    const int buf = t & 1;
    stageB(t, buf);
    if (t + 1 < nt) {
      stageA(t + 1, buf ^ 1);   // overlap next tile's async copies
      // current tile's copies done when only the newest ACHUNKS remain
      asm volatile("s_wait_asynccnt %0" :: "i"(ACHUNKS) : "memory");
    } else {
      asm volatile("s_wait_asynccnt 0" ::: "memory");
    }
    __syncthreads();

    // ---- build fragments per ISA 16-bit A(16x32)/B(32x16) layouts ----
    U16 af[FM], bf[FN];
#pragma unroll
    for (int i = 0; i < FM; ++i) {
      const int m = wr * (FM * 16) + i * 16 + l16;
      af[i].h[0] = *(const h8*)&sA[buf][m][g * 8];        // K = g*8 + 0..7
      af[i].h[1] = *(const h8*)&sA[buf][m][g * 8 + 16];   // K = g*8 + 16..23
    }
#pragma unroll
    for (int j = 0; j < FN; ++j) {
      const int n = wc * (FN * 16) + j * 16 + l16;
      bf[j].h[0] = *(const h8*)&sB[buf][n][g * 8];        // K = g*8 + 0..7
      bf[j].h[1] = *(const h8*)&sB[buf][n][g * 8 + 16];   // K = g*8 + 16..23
    }
#pragma unroll
    for (int i = 0; i < FM; ++i)
#pragma unroll
      for (int j = 0; j < FN; ++j)
        acc[i][j] = __builtin_amdgcn_wmma_f32_16x16x32_f16(
            false, af[i].v, false, bf[j].v, (short)0, acc[i][j], false, false);
    __syncthreads();
  }

  // ---- epilogue ----
  float*  Cfb = Cf ? Cf + (size_t)((long long)blockIdx.z * strideC) : nullptr;
  half_t* Chb = Ch ? Ch + (size_t)((long long)blockIdx.z * strideC) : nullptr;
#pragma unroll
  for (int i = 0; i < FM; ++i)
#pragma unroll
    for (int j = 0; j < FN; ++j)
#pragma unroll
      for (int r = 0; r < 8; ++r) {
        const int m = bm0 + wr * (FM * 16) + i * 16 + g * 8 + r;
        const int n = bn0 + wc * (FN * 16) + j * 16 + l16;
        if (m < M && n < N) {
          float v = alpha * acc[i][j][r];
          if (bias)  v += bias[n];
          if (GELU)  v = gelu_f(v);
          if (resid) v += resid[(size_t)m * ldr + n];
          if (Cfb) Cfb[(size_t)m * ldc + n] = v;
          if (Chb) Chb[(size_t)m * ldc + n] = (half_t)v;
        }
      }
}

// ---------------------------------------------------------------------------
// Weight convert + transpose: W f32 [N,K] (out,in) -> Wt f16 [K,N]
// ---------------------------------------------------------------------------
__global__ void cvt_w_kernel(const float* __restrict__ W, half_t* __restrict__ Wt,
                             int N, int K) {
  const size_t total = (size_t)N * K;
  for (size_t i = (size_t)blockIdx.x * 256 + threadIdx.x; i < total;
       i += (size_t)gridDim.x * 256) {
    const size_t n = i / K, k = i % K;
    Wt[k * N + n] = (half_t)W[i];
  }
}

// ---------------------------------------------------------------------------
// LayerNorm over C=768 rows -> f16 output.  One block per row.
// ---------------------------------------------------------------------------
__global__ __launch_bounds__(256) void ln_kernel(
    const float* __restrict__ x, const float* __restrict__ w,
    const float* __restrict__ b, half_t* __restrict__ out) {
  __shared__ float s1[256], s2[256];
  const int tid = threadIdx.x;
  const float* row = x + (size_t)blockIdx.x * DIMC;
  const float a0 = row[tid], a1 = row[tid + 256], a2 = row[tid + 512];
  s1[tid] = a0 + a1 + a2;
  s2[tid] = a0 * a0 + a1 * a1 + a2 * a2;
  __syncthreads();
  for (int off = 128; off > 0; off >>= 1) {
    if (tid < off) { s1[tid] += s1[tid + off]; s2[tid] += s2[tid + off]; }
    __syncthreads();
  }
  const float mu   = s1[0] * (1.0f / DIMC);
  const float var  = s2[0] * (1.0f / DIMC) - mu * mu;
  const float rstd = rsqrtf(var + 1e-5f);
  half_t* o = out + (size_t)blockIdx.x * DIMC;
  o[tid]       = (half_t)((a0 - mu) * rstd * w[tid]       + b[tid]);
  o[tid + 256] = (half_t)((a1 - mu) * rstd * w[tid + 256] + b[tid + 256]);
  o[tid + 512] = (half_t)((a2 - mu) * rstd * w[tid + 512] + b[tid + 512]);
}

// ---------------------------------------------------------------------------
// Fused LayerNorm + window partition (pad 64->70, 14x14 windows) -> f16.
// grid = (196, 125); one block per windowed token.
// ---------------------------------------------------------------------------
__global__ __launch_bounds__(256) void window_ln_kernel(
    const float* __restrict__ x, const float* __restrict__ w,
    const float* __restrict__ b, half_t* __restrict__ Xw) {
  const int t  = blockIdx.x;
  const int bw = blockIdx.y;
  const int tid = threadIdx.x;
  const int img = bw / 25, wy = (bw % 25) / 5, wx = bw % 5;
  const int i = t / WIN, j = t % WIN;
  const int gy = wy * WIN + i, gx = wx * WIN + j;
  half_t* o = Xw + ((size_t)bw * NP + t) * DIMC;
  if (gy >= HWDIM || gx >= HWDIM) {
    o[tid] = (half_t)0.0f; o[tid + 256] = (half_t)0.0f; o[tid + 512] = (half_t)0.0f;
    return;
  }
  __shared__ float s1[256], s2[256];
  const float* row = x + (((size_t)img * HWDIM + gy) * HWDIM + gx) * DIMC;
  const float a0 = row[tid], a1 = row[tid + 256], a2 = row[tid + 512];
  s1[tid] = a0 + a1 + a2;
  s2[tid] = a0 * a0 + a1 * a1 + a2 * a2;
  __syncthreads();
  for (int off = 128; off > 0; off >>= 1) {
    if (tid < off) { s1[tid] += s1[tid + off]; s2[tid] += s2[tid + off]; }
    __syncthreads();
  }
  const float mu   = s1[0] * (1.0f / DIMC);
  const float var  = s2[0] * (1.0f / DIMC) - mu * mu;
  const float rstd = rsqrtf(var + 1e-5f);
  o[tid]       = (half_t)((a0 - mu) * rstd * w[tid]       + b[tid]);
  o[tid + 256] = (half_t)((a1 - mu) * rstd * w[tid + 256] + b[tid + 256]);
  o[tid + 512] = (half_t)((a2 - mu) * rstd * w[tid + 512] + b[tid + 512]);
}

// ---------------------------------------------------------------------------
// im2col for 3x3x3 SAME conv over [5,64,64,192] f16 -> [T, 5184] f16
// K order = i*27 + kd*9 + kh*3 + kw (matches OIDHW weight flatten).
// ---------------------------------------------------------------------------
__global__ __launch_bounds__(256) void im2col_kernel(
    const half_t* __restrict__ hid, half_t* __restrict__ col) {
  const int t   = blockIdx.x;
  const int d   = t >> 12;
  const int rem = t & 4095;
  const int hh0 = rem >> 6;
  const int ww0 = rem & 63;
  half_t* out = col + (size_t)t * KCONV;
  for (int k = threadIdx.x; k < KCONV; k += 256) {
    const int ci = k / 27, r = k % 27;
    const int kd = r / 9, kh = (r % 9) / 3, kw = r % 3;
    const int dd = d + kd - 1, hh = hh0 + kh - 1, ww = ww0 + kw - 1;
    half_t v = (half_t)0.0f;
    if (dd >= 0 && dd < DDEP && hh >= 0 && hh < HWDIM && ww >= 0 && ww < HWDIM)
      v = hid[(((size_t)dd * HWDIM + hh) * HWDIM + ww) * ACH + ci];
    out[k] = v;
  }
}

// ---------------------------------------------------------------------------
// qkv f32 [24500,2304] (packed [q|k|v]) -> per-(window,head)
// q f16 [1500,196,64], k^T f16 [1500,64,196], v f16 [1500,224,64]
// (v rows 196..223 zeroed).  grid = (12, 125)
// ---------------------------------------------------------------------------
__global__ __launch_bounds__(256) void qkvprep_kernel(
    const float* __restrict__ qkv, half_t* __restrict__ qh,
    half_t* __restrict__ khT, half_t* __restrict__ vh) {
  const int h  = blockIdx.x;
  const int b  = blockIdx.y;
  const int bh = b * HEADS + h;
  for (int idx = threadIdx.x; idx < NP * HDIM; idx += 256) {
    const int t = idx >> 6, c = idx & 63;
    const float* src = qkv + ((size_t)b * NP + t) * (3 * DIMC) + h * HDIM + c;
    qh [((size_t)bh * NP   + t) * HDIM + c] = (half_t)src[0];
    khT[((size_t)bh * HDIM + c) * NP   + t] = (half_t)src[DIMC];
    vh [((size_t)bh * NPAD + t) * HDIM + c] = (half_t)src[2 * DIMC];
  }
  for (int idx = threadIdx.x; idx < (NPAD - NP) * HDIM; idx += 256) {
    const int t = NP + (idx >> 6), c = idx & 63;
    vh[((size_t)bh * NPAD + t) * HDIM + c] = (half_t)0.0f;
  }
}

// ---------------------------------------------------------------------------
// Softmax with fused decomposed rel-pos bias.  grid = (196, 1500).
// dh[k1] = q[t]·rel_h[i-k1+13], dw[k2] = q[t]·rel_w[j-k2+13] (unscaled q).
// Writes padded prob rows [224] (tail zero) as f16.
// ---------------------------------------------------------------------------
__global__ __launch_bounds__(256) void softmax_kernel(
    const float* __restrict__ scores, const half_t* __restrict__ qh,
    const float* __restrict__ rel_h, const float* __restrict__ rel_w,
    half_t* __restrict__ probs) {
  const int t  = blockIdx.x;
  const int bh = blockIdx.y;
  const int i  = t / WIN, j = t % WIN;
  __shared__ float dh[WIN], dw[WIN], red[256];
  const int tid = threadIdx.x;
  if (tid < 2 * WIN) {
    const int k1 = tid % WIN;
    const float* R = (tid < WIN) ? rel_h : rel_w;
    const int pos = ((tid < WIN) ? i : j) - k1 + (WIN - 1);
    const half_t* q = qh + ((size_t)bh * NP + t) * HDIM;
    float s = 0.0f;
    for (int c = 0; c < HDIM; ++c) s += (float)q[c] * R[pos * HDIM + c];
    if (tid < WIN) dh[k1] = s; else dw[k1] = s;
  }
  __syncthreads();
  const float* srow = scores + ((size_t)bh * NP + t) * NP;
  float val = -1e30f;
  if (tid < NP) val = srow[tid] + dh[tid / WIN] + dw[tid % WIN];
  red[tid] = val;
  __syncthreads();
  for (int off = 128; off > 0; off >>= 1) {
    if (tid < off) red[tid] = fmaxf(red[tid], red[tid + off]);
    __syncthreads();
  }
  const float mx = red[0];
  __syncthreads();
  const float e = (tid < NP) ? expf(val - mx) : 0.0f;
  red[tid] = e;
  __syncthreads();
  for (int off = 128; off > 0; off >>= 1) {
    if (tid < off) red[tid] += red[tid + off];
    __syncthreads();
  }
  const float inv = 1.0f / red[0];
  half_t* prow = probs + ((size_t)bh * NP + t) * NPAD;
  if (tid < NPAD) prow[tid] = (tid < NP) ? (half_t)(e * inv) : (half_t)0.0f;
}

// ---------------------------------------------------------------------------
// attn output [1500,196,64] f32 -> merged heads f16 [24500, 768]
// ---------------------------------------------------------------------------
__global__ __launch_bounds__(256) void reorder_kernel(
    const float* __restrict__ ao, half_t* __restrict__ Oh) {
  const size_t total = (size_t)NBH * NP * HDIM;
  for (size_t idx = (size_t)blockIdx.x * 256 + threadIdx.x; idx < total;
       idx += (size_t)gridDim.x * 256) {
    const int c   = (int)(idx & 63);
    const size_t tmp = idx >> 6;
    const int t   = (int)(tmp % NP);
    const int bh2 = (int)(tmp / NP);
    const int b = bh2 / HEADS, h = bh2 % HEADS;
    Oh[((size_t)b * NP + t) * DIMC + h * HDIM + c] = (half_t)ao[idx];
  }
}

// ---------------------------------------------------------------------------
// Window un-partition + residual: x2 = shortcut + unpart(proj_out)
// ---------------------------------------------------------------------------
__global__ __launch_bounds__(256) void unpart_kernel(
    const float* __restrict__ shortcut, const float* __restrict__ proj,
    float* __restrict__ xout) {
  const size_t total = (size_t)T_TOK * DIMC;
  for (size_t idx = (size_t)blockIdx.x * 256 + threadIdx.x; idx < total;
       idx += (size_t)gridDim.x * 256) {
    const int c = (int)(idx % DIMC);
    const size_t tok = idx / DIMC;
    const int gx  = (int)(tok % HWDIM);
    const int gy  = (int)((tok / HWDIM) % HWDIM);
    const int img = (int)(tok / (HWDIM * HWDIM));
    const int bw = img * 25 + (gy / WIN) * 5 + (gx / WIN);
    const int tt = (gy % WIN) * WIN + (gx % WIN);
    xout[idx] = shortcut[idx] + proj[((size_t)bw * NP + tt) * DIMC + c];
  }
}

// ---------------------------------------------------------------------------
// Host-side GEMM dispatch.
// ---------------------------------------------------------------------------
static void launch_gemm(hipStream_t st, bool smallN, bool gelu,
                        const half_t* A, const half_t* B, const float* bias,
                        const float* resid, int ldr, float alpha, float* Cf,
                        half_t* Ch, int ldc, int M, int N, int K,
                        int batch, long long sA, long long sB, long long sC) {
  dim3 blk(256);
  if (smallN) {
    dim3 grid((N + 31) / 32, (M + 127) / 128, batch);
    if (gelu)
      wmma_gemm<128, 32, 8, 1, true><<<grid, blk, 0, st>>>(
          A, B, bias, resid, ldr, alpha, Cf, Ch, ldc, M, N, K, sA, sB, sC);
    else
      wmma_gemm<128, 32, 8, 1, false><<<grid, blk, 0, st>>>(
          A, B, bias, resid, ldr, alpha, Cf, Ch, ldc, M, N, K, sA, sB, sC);
  } else {
    dim3 grid((N + 63) / 64, (M + 127) / 128, batch);
    if (gelu)
      wmma_gemm<128, 64, 4, 2, true><<<grid, blk, 0, st>>>(
          A, B, bias, resid, ldr, alpha, Cf, Ch, ldc, M, N, K, sA, sB, sC);
    else
      wmma_gemm<128, 64, 4, 2, false><<<grid, blk, 0, st>>>(
          A, B, bias, resid, ldr, alpha, Cf, Ch, ldc, M, N, K, sA, sB, sC);
  }
}

extern "C" void kernel_launch(void* const* d_in, const int* in_sizes, int n_in,
                              void* d_out, int out_size, void* d_ws, size_t ws_size,
                              hipStream_t stream) {
  (void)in_sizes; (void)n_in; (void)out_size; (void)ws_size;
  const float* x      = (const float*)d_in[0];
  const float* an1_w  = (const float*)d_in[1];
  const float* an1_b  = (const float*)d_in[2];
  const float* ad1_W  = (const float*)d_in[3];
  const float* ad1_b  = (const float*)d_in[4];
  const float* ac1_W  = (const float*)d_in[5];
  const float* ac1_b  = (const float*)d_in[6];
  const float* au1_W  = (const float*)d_in[7];
  const float* au1_b  = (const float*)d_in[8];
  const float* an2_w  = (const float*)d_in[9];
  const float* an2_b  = (const float*)d_in[10];
  const float* ad2_W  = (const float*)d_in[11];
  const float* ad2_b  = (const float*)d_in[12];
  const float* ac2_W  = (const float*)d_in[13];
  const float* ac2_b  = (const float*)d_in[14];
  const float* au2_W  = (const float*)d_in[15];
  const float* au2_b  = (const float*)d_in[16];
  const float* n1_w   = (const float*)d_in[17];
  const float* n1_b   = (const float*)d_in[18];
  const float* qkv_W  = (const float*)d_in[19];
  const float* qkv_b  = (const float*)d_in[20];
  const float* facu_W = (const float*)d_in[21];
  const float* facv_W = (const float*)d_in[22];
  const float* qfac_W = (const float*)d_in[23];
  const float* vfac_W = (const float*)d_in[24];
  const float* rel_h  = (const float*)d_in[25];
  const float* rel_w  = (const float*)d_in[26];
  const float* proj_W = (const float*)d_in[27];
  const float* proj_b = (const float*)d_in[28];
  const float* n2_w   = (const float*)d_in[29];
  const float* n2_b   = (const float*)d_in[30];
  const float* mlp_W1 = (const float*)d_in[31];
  const float* mlp_b1 = (const float*)d_in[32];
  const float* mlp_W2 = (const float*)d_in[33];
  const float* mlp_b2 = (const float*)d_in[34];
  float* out = (float*)d_out;

  // ---- workspace carve-out (≈1.5 GB; MI455X has 432 GiB) ----
  char* base = (char*)d_ws;
  size_t off = 0;
  auto alloc = [&](size_t bytes) -> void* {
    void* p = base + off;
    off += (bytes + 255) & ~(size_t)255;
    return p;
  };
  half_t* yh     = (half_t*)alloc((size_t)T_TOK * DIMC * 2);
  half_t* hid    = (half_t*)alloc((size_t)T_TOK * ACH * 2);
  half_t* colb   = (half_t*)alloc((size_t)T_TOK * KCONV * 2);
  half_t* gbuf   = (half_t*)alloc((size_t)T_TOK * ACH * 2);
  float*  x1     = (float*) alloc((size_t)T_TOK * DIMC * 4);
  float*  x2     = (float*) alloc((size_t)T_TOK * DIMC * 4);
  float*  x3     = (float*) alloc((size_t)T_TOK * DIMC * 4);
  half_t* Xw     = (half_t*)alloc((size_t)TW_TOK * DIMC * 2);
  float*  qkvb   = (float*) alloc((size_t)TW_TOK * 3 * DIMC * 4);
  half_t* fuh    = (half_t*)alloc((size_t)TW_TOK * RLO * 2);
  half_t* t1q    = (half_t*)alloc((size_t)TW_TOK * RLO * 2);
  half_t* t1v    = (half_t*)alloc((size_t)TW_TOK * RLO * 2);
  half_t* qh     = (half_t*)alloc((size_t)NBH * NP * HDIM * 2);
  half_t* khT    = (half_t*)alloc((size_t)NBH * HDIM * NP * 2);
  half_t* vh     = (half_t*)alloc((size_t)NBH * NPAD * HDIM * 2);
  float*  scores = (float*) alloc((size_t)NBH * NP * NP * 4);
  half_t* probs  = (half_t*)alloc((size_t)NBH * NP * NPAD * 2);
  float*  attno  = (float*) alloc((size_t)NBH * NP * HDIM * 4);
  half_t* Oh     = (half_t*)alloc((size_t)TW_TOK * DIMC * 2);
  float*  projo  = (float*) alloc((size_t)TW_TOK * DIMC * 4);
  half_t* mlph   = (half_t*)alloc((size_t)T_TOK * 4 * DIMC * 2);
  // f16 transposed weights
  half_t* ad1Wt  = (half_t*)alloc((size_t)DIMC * ACH * 2);
  half_t* Wc1t   = (half_t*)alloc((size_t)KCONV * ACH * 2);
  half_t* au1Wt  = (half_t*)alloc((size_t)ACH * DIMC * 2);
  half_t* ad2Wt  = (half_t*)alloc((size_t)DIMC * ACH * 2);
  half_t* Wc2t   = (half_t*)alloc((size_t)KCONV * ACH * 2);
  half_t* au2Wt  = (half_t*)alloc((size_t)ACH * DIMC * 2);
  half_t* qkvWt  = (half_t*)alloc((size_t)DIMC * 3 * DIMC * 2);
  half_t* facuWt = (half_t*)alloc((size_t)DIMC * RLO * 2);
  half_t* qfacWt = (half_t*)alloc((size_t)RLO * RLO * 2);
  half_t* vfacWt = (half_t*)alloc((size_t)RLO * RLO * 2);
  half_t* facvWt = (half_t*)alloc((size_t)RLO * DIMC * 2);
  half_t* projWt = (half_t*)alloc((size_t)DIMC * DIMC * 2);
  half_t* mlpW1t = (half_t*)alloc((size_t)DIMC * 4 * DIMC * 2);
  half_t* mlpW2t = (half_t*)alloc((size_t)4 * DIMC * DIMC * 2);

  auto cvt = [&](const float* W, half_t* Wt, int N, int K) {
    const size_t total = (size_t)N * K;
    unsigned blocks = (unsigned)((total + 255) / 256);
    if (blocks > 16384u) blocks = 16384u;
    cvt_w_kernel<<<blocks, 256, 0, stream>>>(W, Wt, N, K);
  };
  cvt(ad1_W, ad1Wt, ACH, DIMC);
  cvt(ac1_W, Wc1t, ACH, KCONV);
  cvt(au1_W, au1Wt, DIMC, ACH);
  cvt(ad2_W, ad2Wt, ACH, DIMC);
  cvt(ac2_W, Wc2t, ACH, KCONV);
  cvt(au2_W, au2Wt, DIMC, ACH);
  cvt(qkv_W, qkvWt, 3 * DIMC, DIMC);
  cvt(facu_W, facuWt, RLO, DIMC);
  cvt(qfac_W, qfacWt, RLO, RLO);
  cvt(vfac_W, vfacWt, RLO, RLO);
  cvt(facv_W, facvWt, DIMC, RLO);
  cvt(proj_W, projWt, DIMC, DIMC);
  cvt(mlp_W1, mlpW1t, 4 * DIMC, DIMC);
  cvt(mlp_W2, mlpW2t, DIMC, 4 * DIMC);

  const float SCALE = 0.125f;  // 64^-0.5
  const float S     = 1.0f;

  // ================= adapter 1 : x -> x1 =================
  ln_kernel<<<T_TOK, 256, 0, stream>>>(x, an1_w, an1_b, yh);
  launch_gemm(stream, false, false, yh, ad1Wt, ad1_b, nullptr, 0, 1.0f,
              nullptr, hid, ACH, T_TOK, ACH, DIMC, 1, 0, 0, 0);
  im2col_kernel<<<T_TOK, 256, 0, stream>>>(hid, colb);
  launch_gemm(stream, false, true, colb, Wc1t, ac1_b, nullptr, 0, 1.0f,
              nullptr, gbuf, ACH, T_TOK, ACH, KCONV, 1, 0, 0, 0);
  launch_gemm(stream, false, false, gbuf, au1Wt, au1_b, x, DIMC, 1.0f,
              x1, nullptr, DIMC, T_TOK, DIMC, ACH, 1, 0, 0, 0);

  // ================= attention : x1 -> x2 =================
  window_ln_kernel<<<dim3(NP, NWIN), 256, 0, stream>>>(x1, n1_w, n1_b, Xw);
  launch_gemm(stream, false, false, Xw, qkvWt, qkv_b, nullptr, 0, 1.0f,
              qkvb, nullptr, 3 * DIMC, TW_TOK, 3 * DIMC, DIMC, 1, 0, 0, 0);
  // LoRA-style factor path: fu = Xw@facu^T ; new_q/new_v folded into qkv
  launch_gemm(stream, true, false, Xw, facuWt, nullptr, nullptr, 0, 1.0f,
              nullptr, fuh, RLO, TW_TOK, RLO, DIMC, 1, 0, 0, 0);
  launch_gemm(stream, true, false, fuh, qfacWt, nullptr, nullptr, 0, 1.0f,
              nullptr, t1q, RLO, TW_TOK, RLO, RLO, 1, 0, 0, 0);
  launch_gemm(stream, true, false, fuh, vfacWt, nullptr, nullptr, 0, 1.0f,
              nullptr, t1v, RLO, TW_TOK, RLO, RLO, 1, 0, 0, 0);
  launch_gemm(stream, false, false, t1q, facvWt, nullptr, qkvb, 3 * DIMC, S,
              qkvb, nullptr, 3 * DIMC, TW_TOK, DIMC, RLO, 1, 0, 0, 0);
  launch_gemm(stream, false, false, t1v, facvWt, nullptr, qkvb + 2 * DIMC,
              3 * DIMC, S, qkvb + 2 * DIMC, nullptr, 3 * DIMC,
              TW_TOK, DIMC, RLO, 1, 0, 0, 0);
  qkvprep_kernel<<<dim3(HEADS, NWIN), 256, 0, stream>>>(qkvb, qh, khT, vh);
  // scores = (q*SCALE) @ k^T   batched over 1500 (window,head)
  launch_gemm(stream, false, false, qh, khT, nullptr, nullptr, 0, SCALE,
              scores, nullptr, NP, NP, NP, HDIM, NBH,
              (long long)NP * HDIM, (long long)HDIM * NP, (long long)NP * NP);
  softmax_kernel<<<dim3(NP, NBH), 256, 0, stream>>>(scores, qh, rel_h, rel_w, probs);
  // out = probs @ v            batched, K padded to 224
  launch_gemm(stream, false, false, probs, vh, nullptr, nullptr, 0, 1.0f,
              attno, nullptr, HDIM, NP, HDIM, NPAD, NBH,
              (long long)NP * NPAD, (long long)NPAD * HDIM, (long long)NP * HDIM);
  reorder_kernel<<<8192, 256, 0, stream>>>(attno, Oh);
  launch_gemm(stream, false, false, Oh, projWt, proj_b, nullptr, 0, 1.0f,
              projo, nullptr, DIMC, TW_TOK, DIMC, DIMC, 1, 0, 0, 0);
  unpart_kernel<<<8192, 256, 0, stream>>>(x1, projo, x2);

  // ================= adapter 2 : x2 -> x3 =================
  ln_kernel<<<T_TOK, 256, 0, stream>>>(x2, an2_w, an2_b, yh);
  launch_gemm(stream, false, false, yh, ad2Wt, ad2_b, nullptr, 0, 1.0f,
              nullptr, hid, ACH, T_TOK, ACH, DIMC, 1, 0, 0, 0);
  im2col_kernel<<<T_TOK, 256, 0, stream>>>(hid, colb);
  launch_gemm(stream, false, true, colb, Wc2t, ac2_b, nullptr, 0, 1.0f,
              nullptr, gbuf, ACH, T_TOK, ACH, KCONV, 1, 0, 0, 0);
  launch_gemm(stream, false, false, gbuf, au2Wt, au2_b, x2, DIMC, 1.0f,
              x3, nullptr, DIMC, T_TOK, DIMC, ACH, 1, 0, 0, 0);

  // ================= MLP : out = x3 + W2(gelu(W1 ln(x3))) =================
  ln_kernel<<<T_TOK, 256, 0, stream>>>(x3, n2_w, n2_b, yh);
  launch_gemm(stream, false, true, yh, mlpW1t, mlp_b1, nullptr, 0, 1.0f,
              nullptr, mlph, 4 * DIMC, T_TOK, 4 * DIMC, DIMC, 1, 0, 0, 0);
  launch_gemm(stream, false, false, mlph, mlpW2t, mlp_b2, x3, DIMC, 1.0f,
              out, nullptr, DIMC, T_TOK, DIMC, 4 * DIMC, 1, 0, 0, 0);
}
